// Transformer_29832842838303
// MI455X (gfx1250) — matrix operand order repaired and checked by reading, not compiled, not run on hardware
//
#include <hip/hip_runtime.h>
#include <hip/hip_bf16.h>

#define DIMN 128

typedef __attribute__((ext_vector_type(16))) __bf16 v16bf;
typedef __attribute__((ext_vector_type(8)))  __bf16 v8bf;
typedef __attribute__((ext_vector_type(8)))  float  v8f;

// ---------------------------------------------------------------------------
// wave32 full reduction (all lanes get the sum)
// ---------------------------------------------------------------------------
__device__ __forceinline__ float wave_sum(float v) {
#pragma unroll
    for (int m = 16; m > 0; m >>= 1) v += __shfl_xor(v, m, 32);
    return v;
}

// ---------------------------------------------------------------------------
// WMMA operand loaders from pre-converted bf16 buffers.
// A 16x32 bf16 order used here (contiguous, not the ISA table order): lane L -> row M=L&15,
//   lanes 0-15: halves 0..15 = K 0..15
//   lanes16-31: halves 0..15 = K 16..31
// rp points at (row base + kbase); two 16B chunks at +k0 and +k0+8.
// ---------------------------------------------------------------------------
__device__ __forceinline__ v16bf load_a_bf(const __bf16* rp, int lane) {
    const int k0 = (lane >> 4) << 4;   // 0 or 16
    const v8bf lo = *(const v8bf*)(rp + k0);
    const v8bf hi = *(const v8bf*)(rp + k0 + 8);
    v16bf r;
#pragma unroll
    for (int i = 0; i < 8; ++i) { r[i] = lo[i]; r[8 + i] = hi[i]; }
    return r;
}

// B 32x16 bf16 layout: lane L -> col N=L&15,
//   lanes 0-15 hold K 0..15, lanes 16-31 hold K 16..31 (halves = consecutive K)
// B column contiguous in memory (bf16 emb row == column of emb^T): one 32B load.
__device__ __forceinline__ v16bf load_b_contig_bf(const __bf16* rp, int kbase, int lane) {
    return *(const v16bf*)(rp + kbase + ((lane >> 4) << 4));
}

// ---------------------------------------------------------------------------
// One-time conversions: f32 -> bf16 (grid-stride)
// ---------------------------------------------------------------------------
__global__ void k_cvt_bf16(const float* __restrict__ src, __bf16* __restrict__ dst,
                           long long n) {
    long long i = (long long)blockIdx.x * blockDim.x + threadIdx.x;
    const long long stride = (long long)gridDim.x * blockDim.x;
    for (; i < n; i += stride) dst[i] = (__bf16)src[i];
}

// One-time swizzle of Wq[256][128] f32 into WMMA-B bf16 layout:
//   wqsw[((kk*2 + g)*DIMN + n)*16 + i] = Wq[kk*32 + g*16 + i][n]
// so a lane's 16 B-halves for k-step kk are one contiguous 32B vector.
__global__ void k_swizzle_wq(const float* __restrict__ Wq, __bf16* __restrict__ wqsw) {
    const int v = blockIdx.x * blockDim.x + threadIdx.x;   // 0..(8*2*128)-1
    if (v >= 8 * 2 * DIMN) return;
    const int n  = v & (DIMN - 1);
    const int g  = (v >> 7) & 1;
    const int kk = v >> 8;
    __bf16* out = wqsw + (size_t)v * 16;
#pragma unroll
    for (int i = 0; i < 16; ++i)
        out[i] = (__bf16)Wq[(size_t)(kk * 32 + g * 16 + i) * DIMN + n];
}

// ---------------------------------------------------------------------------
// K1: x = emb[iid]*sqrt(128); nx = LN(x, ln1_w, ln1_b).  One wave per row.
// ---------------------------------------------------------------------------
__global__ void k_embed_ln(const int* __restrict__ iid, const float* __restrict__ emb,
                           const float* __restrict__ w, const float* __restrict__ b,
                           float* __restrict__ nx, int Ni) {
    const int row  = (blockIdx.x * blockDim.x + threadIdx.x) >> 5;
    const int lane = threadIdx.x & 31;
    if (row >= Ni) return;
    const float s = 11.313708498984761f;  // sqrt(128)
    float4 x = ((const float4*)(emb + (size_t)iid[row] * DIMN))[lane];
    x.x *= s; x.y *= s; x.z *= s; x.w *= s;
    const float m = wave_sum(x.x + x.y + x.z + x.w) * (1.0f / DIMN);
    const float dx = x.x - m, dy = x.y - m, dz = x.z - m, dw = x.w - m;
    const float var = wave_sum(dx * dx + dy * dy + dz * dz + dw * dw) * (1.0f / DIMN);
    const float r = rsqrtf(var + 1e-5f);
    const float4 wv = ((const float4*)w)[lane];
    const float4 bv = ((const float4*)b)[lane];
    float4 o;
    o.x = dx * r * wv.x + bv.x;
    o.y = dy * r * wv.y + bv.y;
    o.z = dz * r * wv.z + bv.z;
    o.w = dw * r * wv.w + bv.w;
    ((float4*)(nx + (size_t)row * DIMN))[lane] = o;
}

// ---------------------------------------------------------------------------
// K2: graph attention (dst = e % Ni -> node i owns edges e = i + k*Ni),
//     then x2 = LN(x + wv/z, enc_norm).  Writes f32 x2 and bf16 x2 (GEMM A).
// ---------------------------------------------------------------------------
__global__ void k_attn(const int* __restrict__ iid, const int* __restrict__ isrc,
                       const float* __restrict__ emb, const float* __restrict__ nx,
                       const float* __restrict__ w, const float* __restrict__ b,
                       float* __restrict__ x2, __bf16* __restrict__ x2bf,
                       int Ni, int E) {
    const int node = (blockIdx.x * blockDim.x + threadIdx.x) >> 5;
    const int lane = threadIdx.x & 31;
    if (node >= Ni) return;
    const float4 q = ((const float4*)(nx + (size_t)node * DIMN))[lane];
    float4 wv = make_float4(0.f, 0.f, 0.f, 0.f);
    float z = 0.f;
    const float inv_sqrt_dk = 0.08838834764831845f;  // 1/sqrt(128)
    for (int e = node; e < E; e += Ni) {
        const int s = isrc[e];
        const float4 kv = ((const float4*)(nx + (size_t)s * DIMN))[lane];
        float d = wave_sum(q.x * kv.x + q.y * kv.y + q.z * kv.z + q.w * kv.w);
        d = fminf(10.f, fmaxf(-10.f, d * inv_sqrt_dk));
        const float sc = expf(d);
        wv.x += kv.x * sc; wv.y += kv.y * sc; wv.z += kv.z * sc; wv.w += kv.w * sc;
        z += sc;
    }
    const float iz = 1.0f / z;
    const float s = 11.313708498984761f;
    const float4 x = ((const float4*)(emb + (size_t)iid[node] * DIMN))[lane];
    float4 y;
    y.x = x.x * s + wv.x * iz;
    y.y = x.y * s + wv.y * iz;
    y.z = x.z * s + wv.z * iz;
    y.w = x.w * s + wv.w * iz;
    const float m = wave_sum(y.x + y.y + y.z + y.w) * (1.0f / DIMN);
    const float dx = y.x - m, dy = y.y - m, dz = y.z - m, dw = y.w - m;
    const float var = wave_sum(dx * dx + dy * dy + dz * dz + dw * dw) * (1.0f / DIMN);
    const float r = rsqrtf(var + 1e-5f);
    const float4 wvn = ((const float4*)w)[lane];
    const float4 bvn = ((const float4*)b)[lane];
    float4 o;
    o.x = dx * r * wvn.x + bvn.x;
    o.y = dy * r * wvn.y + bvn.y;
    o.z = dz * r * wvn.z + bvn.z;
    o.w = dw * r * wvn.w + bvn.w;
    ((float4*)(x2 + (size_t)node * DIMN))[lane] = o;
    __bf16* brow = x2bf + (size_t)node * DIMN + lane * 4;
    brow[0] = (__bf16)o.x; brow[1] = (__bf16)o.y;
    brow[2] = (__bf16)o.z; brow[3] = (__bf16)o.w;
}

// ---------------------------------------------------------------------------
// K3: per-target mean of gathered ft (adst = e % Nt).  One wave per target.
// ---------------------------------------------------------------------------
__global__ void k_mean(const int* __restrict__ asrc, const float* __restrict__ x2,
                       float* __restrict__ mean, int Nt, int Ea) {
    const int t = (blockIdx.x * blockDim.x + threadIdx.x) >> 5;
    const int lane = threadIdx.x & 31;
    if (t >= Nt) return;
    float4 acc = make_float4(0.f, 0.f, 0.f, 0.f);
    float cnt = 0.f;
    for (int e = t; e < Ea; e += Nt) {
        const float4 v = ((const float4*)(x2 + (size_t)asrc[e] * DIMN))[lane];
        acc.x += v.x; acc.y += v.y; acc.z += v.z; acc.w += v.w;
        cnt += 1.f;
    }
    const float ic = 1.0f / fmaxf(cnt, 1.0f);
    acc.x *= ic; acc.y *= ic; acc.z *= ic; acc.w *= ic;
    ((float4*)(mean + (size_t)t * DIMN))[lane] = acc;
}

// ---------------------------------------------------------------------------
// K4: f[t] = concat(target_emb, mean[t]) @ Wr   (small matvec, 128 thr/target)
// ---------------------------------------------------------------------------
__global__ void k_fproj(const float* __restrict__ tgt, const float* __restrict__ mean,
                        const float* __restrict__ Wr, float* __restrict__ f, int Nt) {
    const int t = blockIdx.x;
    const int n = threadIdx.x;
    if (t >= Nt) return;
    float s = 0.f;
    const float* mrow = mean + (size_t)t * DIMN;
#pragma unroll 4
    for (int k = 0; k < DIMN; ++k) s += tgt[k] * Wr[(size_t)k * DIMN + n];
#pragma unroll 4
    for (int k = 0; k < DIMN; ++k) s += mrow[k] * Wr[(size_t)(DIMN + k) * DIMN + n];
    f[(size_t)t * DIMN + n] = s;
}

// ---------------------------------------------------------------------------
// K5: e = tanh(concat(ft_e, pos) @ Wq) via WMMA bf16.
//     Block = 8 waves; block handles one 16-row M tile; wave w -> N tile w.
//     A rows gathered from bf16 x2/pos; B from pre-swizzled bf16 Wq (1 vector
//     load per k-step).  All operands already bf16: no converts in the loop.
// ---------------------------------------------------------------------------
__global__ void k_edge_gemm(const int* __restrict__ asrc, const int* __restrict__ apid,
                            const __bf16* __restrict__ x2bf, const __bf16* __restrict__ posbf,
                            const __bf16* __restrict__ wqsw, float* __restrict__ ebuf,
                            int Ea) {
    const int lane  = threadIdx.x & 31;
    const int w     = threadIdx.x >> 5;          // 0..7 -> N tile
    const int mbase = blockIdx.x * 16;
    const int n     = w * 16 + (lane & 15);
    const int g     = lane >> 4;                 // B half select
    int row = mbase + (lane & 15);
    if (row >= Ea) row = Ea - 1;                 // clamp loads; stores guarded
    const int src = asrc[row];
    const int pid = apid[row];
    const __bf16* ftrow = x2bf  + (size_t)src * DIMN;
    const __bf16* prow  = posbf + (size_t)pid * DIMN;
    v8f acc = {};
#pragma unroll
    for (int kk = 0; kk < 8; ++kk) {
        const int kbase = kk * 32;
        const __bf16* rp = (kbase < DIMN) ? (ftrow + kbase) : (prow + (kbase - DIMN));
        const v16bf a  = load_a_bf(rp, lane);
        const v16bf bm = *(const v16bf*)(wqsw + ((size_t)((kk * 2 + g) * DIMN + n)) * 16);
        acc = __builtin_amdgcn_wmma_f32_16x16x32_bf16(false, a, false, bm,
                                                      (short)0, acc, false, false);
    }
#pragma unroll
    for (int r = 0; r < 8; ++r) {
        const int rr = mbase + r + (g << 3);
        if (rr < Ea) ebuf[(size_t)rr * DIMN + n] = tanhf(acc[r]);
    }
}

// ---------------------------------------------------------------------------
// K6: select[t] = sum_e ft_e * dot(e_vec, f[t]).  One wave per target.
//     Emits select directly in bf16 (only consumer is the WMMA scores GEMM).
// ---------------------------------------------------------------------------
__global__ void k_select(const int* __restrict__ asrc, const float* __restrict__ ebuf,
                         const float* __restrict__ f, const float* __restrict__ x2,
                         __bf16* __restrict__ selbf, int Nt, int Ea) {
    const int t = (blockIdx.x * blockDim.x + threadIdx.x) >> 5;
    const int lane = threadIdx.x & 31;
    if (t >= Nt) return;
    const float4 fv = ((const float4*)(f + (size_t)t * DIMN))[lane];
    float4 acc = make_float4(0.f, 0.f, 0.f, 0.f);
    for (int e = t; e < Ea; e += Nt) {
        const float4 ev = ((const float4*)(ebuf + (size_t)e * DIMN))[lane];
        const float s = wave_sum(ev.x * fv.x + ev.y * fv.y + ev.z * fv.z + ev.w * fv.w);
        const float4 xv = ((const float4*)(x2 + (size_t)asrc[e] * DIMN))[lane];
        acc.x += xv.x * s; acc.y += xv.y * s; acc.z += xv.z * s; acc.w += xv.w * s;
    }
    __bf16* srow = selbf + (size_t)t * DIMN + lane * 4;
    srow[0] = (__bf16)acc.x; srow[1] = (__bf16)acc.y;
    srow[2] = (__bf16)acc.z; srow[3] = (__bf16)acc.w;
}

// ---------------------------------------------------------------------------
// K7: scores = select @ emb[1:].T via WMMA bf16.
//     Block = one 16-col N tile (8 waves); B held in registers across all
//     128 M tiles; A is a 32B vector pair per k-step from bf16 select.
//     Output (409 MB) written with non-temporal stores: streaming, never
//     re-read -> keep the 192 MB L2 for emb/select instead.
// ---------------------------------------------------------------------------
__global__ void k_scores(const __bf16* __restrict__ selbf, const __bf16* __restrict__ embbf,
                         float* __restrict__ out, int Nt, int Nout, int NumNode) {
    const int lane = threadIdx.x & 31;
    const int w    = threadIdx.x >> 5;
    const int nb   = blockIdx.x * 16;
    const int col  = nb + (lane & 15);
    int erow = col + 1;
    if (erow >= NumNode) erow = NumNode - 1;     // clamp loads; stores guarded
    const __bf16* brow = embbf + (size_t)erow * DIMN;
    const v16bf b0 = load_b_contig_bf(brow, 0,  lane);
    const v16bf b1 = load_b_contig_bf(brow, 32, lane);
    const v16bf b2 = load_b_contig_bf(brow, 64, lane);
    const v16bf b3 = load_b_contig_bf(brow, 96, lane);
    const int Mt = (Nt + 15) >> 4;
    for (int mt = w; mt < Mt; mt += 8) {
        int mrow = mt * 16 + (lane & 15);
        if (mrow >= Nt) mrow = Nt - 1;
        const __bf16* arow = selbf + (size_t)mrow * DIMN;
        v8f acc = {};
        acc = __builtin_amdgcn_wmma_f32_16x16x32_bf16(false, load_a_bf(arow,      lane),
                                                      false, b0, (short)0, acc, false, false);
        acc = __builtin_amdgcn_wmma_f32_16x16x32_bf16(false, load_a_bf(arow + 32, lane),
                                                      false, b1, (short)0, acc, false, false);
        acc = __builtin_amdgcn_wmma_f32_16x16x32_bf16(false, load_a_bf(arow + 64, lane),
                                                      false, b2, (short)0, acc, false, false);
        acc = __builtin_amdgcn_wmma_f32_16x16x32_bf16(false, load_a_bf(arow + 96, lane),
                                                      false, b3, (short)0, acc, false, false);
#pragma unroll
        for (int r = 0; r < 8; ++r) {
            const int rr = mt * 16 + r + ((lane >> 4) << 3);
            if (col < Nout && rr < Nt)
                __builtin_nontemporal_store(acc[r], &out[(size_t)rr * Nout + col]);
        }
    }
}

// ---------------------------------------------------------------------------
// Host-side launch
// ---------------------------------------------------------------------------
extern "C" void kernel_launch(void* const* d_in, const int* in_sizes, int n_in,
                              void* d_out, int out_size, void* d_ws, size_t ws_size,
                              hipStream_t stream) {
    const int*   iid   = (const int*)d_in[0];
    const int*   isrc  = (const int*)d_in[1];
    // d_in[2] inter_dst_arange: structure (arange % Ni) exploited directly
    const int*   asrc  = (const int*)d_in[3];
    // d_in[4] agg_dst_arange: structure (arange % Nt) exploited directly
    const int*   apid  = (const int*)d_in[5];
    // d_in[6] n_targets: device scalar; recovered from out_size below
    const float* emb   = (const float*)d_in[7];
    const float* ln1w  = (const float*)d_in[8];
    const float* ln1b  = (const float*)d_in[9];
    const float* encw  = (const float*)d_in[10];
    const float* encb  = (const float*)d_in[11];
    const float* tgt   = (const float*)d_in[12];
    const float* pos   = (const float*)d_in[13];
    const float* Wq    = (const float*)d_in[14];
    const float* Wr    = (const float*)d_in[15];

    const int Ni      = in_sizes[0];
    const int E       = in_sizes[1];
    const int Ea      = in_sizes[3];
    const int Npos    = in_sizes[13];              // 200*128
    const int NumNode = in_sizes[7] / DIMN;
    const int Nout    = NumNode - 1;               // 49999
    const int Nt      = out_size / Nout;           // 2048

    // ---- workspace carve (256B aligned regions) ----
    char* p = (char*)d_ws;
    auto carve = [&p](size_t bytes) -> char* {
        char* r = p;
        p += (bytes + 255) & ~(size_t)255;
        return r;
    };
    const size_t maxNE = (size_t)(Ni > Ea ? Ni : Ea) * DIMN;
    float*  nx    = (float*)carve(maxNE * sizeof(float));      // aliased with ebuf
    float*  ebuf  = nx;                                        // nx dead after K2
    float*  x2    = (float*)carve((size_t)Ni * DIMN * sizeof(float));
    float*  mean  = (float*)carve((size_t)Nt * DIMN * sizeof(float));
    float*  f     = (float*)carve((size_t)Nt * DIMN * sizeof(float));
    __bf16* x2bf  = (__bf16*)carve((size_t)Ni * DIMN * sizeof(__bf16));
    __bf16* embbf = (__bf16*)carve((size_t)NumNode * DIMN * sizeof(__bf16));
    __bf16* posbf = (__bf16*)carve((size_t)Npos * sizeof(__bf16));
    __bf16* wqsw  = (__bf16*)carve((size_t)8 * 2 * DIMN * 16 * sizeof(__bf16));
    __bf16* selbf = (__bf16*)carve((size_t)Nt * DIMN * sizeof(__bf16));

    const int TPB = 256;                           // 8 waves (wave32)
    const int rowsPerBlk = TPB / 32;

    // ---- one-time precision conversions / swizzles (input-only deps) ----
    {
        const long long ne = (long long)NumNode * DIMN;
        k_cvt_bf16<<<dim3(1024), dim3(TPB), 0, stream>>>(emb, embbf, ne);
        k_cvt_bf16<<<dim3(64), dim3(TPB), 0, stream>>>(pos, posbf, (long long)Npos);
        k_swizzle_wq<<<dim3((8 * 2 * DIMN + TPB - 1) / TPB), dim3(TPB), 0, stream>>>(Wq, wqsw);
    }

    k_embed_ln<<<dim3((Ni + rowsPerBlk - 1) / rowsPerBlk), dim3(TPB), 0, stream>>>(
        iid, emb, ln1w, ln1b, nx, Ni);

    k_attn<<<dim3((Ni + rowsPerBlk - 1) / rowsPerBlk), dim3(TPB), 0, stream>>>(
        iid, isrc, emb, nx, encw, encb, x2, x2bf, Ni, E);

    k_mean<<<dim3((Nt + rowsPerBlk - 1) / rowsPerBlk), dim3(TPB), 0, stream>>>(
        asrc, x2, mean, Nt, Ea);

    k_fproj<<<dim3(Nt), dim3(DIMN), 0, stream>>>(tgt, mean, Wr, f, Nt);

    k_edge_gemm<<<dim3((Ea + 15) / 16), dim3(TPB), 0, stream>>>(
        asrc, apid, x2bf, posbf, wqsw, ebuf, Ea);

    k_select<<<dim3((Nt + rowsPerBlk - 1) / rowsPerBlk), dim3(TPB), 0, stream>>>(
        asrc, ebuf, f, x2, selbf, Nt, Ea);

    k_scores<<<dim3((Nout + 15) / 16), dim3(TPB), 0, stream>>>(
        selbf, embbf, (float*)d_out, Nt, Nout, NumNode);
}
